// Encoder_56384330662000
// MI455X (gfx1250) — matrix operand order repaired and checked by reading, not compiled, not run on hardware
//
#include <hip/hip_runtime.h>
#include <hip/hip_bf16.h>

// ---------------------------------------------------------------------------
// Transformer encoder for MI455X (gfx1250, wave32, WMMA + TDM).
// All GEMMs in bf16 via v_wmma_f32_16x16x32_bf16 (f32 accum). Flash-attention
// keeps 16q x 64k score tiles in registers/LDS (online softmax). Dense GEMM
// weight tiles are staged into LDS with the Tensor Data Mover
// (tensor_load_to_lds, TENSORcnt) and double-buffered against WMMA compute.
// ---------------------------------------------------------------------------

typedef __bf16 bf16;
typedef bf16  v16bf __attribute__((ext_vector_type(16)));
typedef bf16  v8bf  __attribute__((ext_vector_type(8)));
typedef float v8f   __attribute__((ext_vector_type(8)));
typedef unsigned int u32x4 __attribute__((ext_vector_type(4)));
typedef int  i32x4 __attribute__((ext_vector_type(4)));
typedef int  i32x8 __attribute__((ext_vector_type(8)));

#define LNUM   6
#define DMODEL 512
#define NHEAD  8
#define HDIM   64
#define FDIM   2048
#define NB     2
#define SEQ    2048
#define NTOK   (NB * SEQ)        // 4096 tokens
#define NROWQ  (NTOK * NHEAD)    // 32768 token-head rows

// ---------------------------------------------------------------------------
// WMMA helpers (CDNA5 16x16x32 bf16 -> f32)
// ---------------------------------------------------------------------------
__device__ __forceinline__ v8f wmma_bf16(v16bf a, v16bf b, v8f c) {
  // (neg_a, A, neg_b, B, c_mod, C, reuse_a, reuse_b)
  return __builtin_amdgcn_wmma_f32_16x16x32_bf16(false, a, false, b, (short)0, c,
                                                 false, false);
}

// A-matrix fragment (16x32, M x K), contiguous K order (matches B frags):
// lane<16 : row M=lane,    halves 0-15 = K[0..15]
// lane>=16: row M=lane-16, halves 0-15 = K[16..31]
__device__ __forceinline__ v16bf load_a_frag(const bf16* p, int lda, int lane) {
  int m  = lane & 15;
  int kb = (lane >> 4) << 4;                 // 0 or 16
  const bf16* q = p + (size_t)m * lda + kb;
  v8bf lo = *(const v8bf*)(q);
  v8bf hi = *(const v8bf*)(q + 8);
  v16bf r;
#pragma unroll
  for (int i = 0; i < 8; ++i) { r[i] = lo[i]; r[i + 8] = hi[i]; }
  return r;
}

// B-matrix fragment (32x16, K x N) from a row-major [N,K] source (torch-style
// weight W[n,k]): lane<16 -> col n=lane, K[0..15]; lane>=16 -> col n=lane-16,
// K[16..31]; one contiguous 32B load per lane.
__device__ __forceinline__ v16bf load_b_frag(const bf16* p, int ldb, int lane) {
  int n  = lane & 15;
  int kb = (lane >> 4) << 4;                 // 0 or 16
  return *(const v16bf*)(p + (size_t)n * ldb + kb);
}

// Same B-fragment pattern reading a 64x32 row-major tile staged in LDS.
__device__ __forceinline__ v16bf load_b_frag_lds(const bf16* p, int lane) {
  int n  = lane & 15;
  int kb = (lane >> 4) << 4;
  return *(const v16bf*)(p + n * 32 + kb);
}

// C/D layout: lane l, vgpr r -> row r + 8*(l>=16), col l&15.

// ---------------------------------------------------------------------------
// Tensor Data Mover: DMA one 64-row x 32-elem bf16 tile (row stride = K elems)
// from global into LDS. D# per CDNA5 ISA ch.8: group0 = {count, lds_addr,
// global_addr, type=2}; group1 = {data_size=2B, dims/tile/strides}; groups
// 2/3 zero (2-D tile). Tracked on TENSORcnt.
// amdgpu-toolchain (clang-23) builtin arity: 6 args
//   (u32x4 g0, i32x8 g1, i32x4, i32x4, i32x8, i32 cpol)
// ---------------------------------------------------------------------------
__device__ __forceinline__ void tdm_load_tile64x32(unsigned lds_off,
                                                   const bf16* gptr, int K) {
  unsigned long long ga = (unsigned long long)(size_t)gptr;
  u32x4 g0;
  g0[0] = 1u;                                         // count=1, user mode
  g0[1] = lds_off;                                    // LDS byte address
  g0[2] = (unsigned)ga;                               // global_addr[31:0]
  g0[3] = (unsigned)((ga >> 32) & 0x01FFFFFFu) | 0x80000000u;  // [56:32]|type=2
  unsigned td0   = (unsigned)K;                       // tensor_dim0 (elems)
  unsigned td1   = 0x007FFFFFu;                       // tensor_dim1 (large)
  unsigned tile0 = 32u, tile1 = 64u;                  // 32 elems x 64 rows
  unsigned long long s0 = (unsigned long long)K;      // dim0 stride (elems)
  i32x8 g1;
  g1[0] = 0x00010000;                                 // data_size=1 (2 bytes)
  g1[1] = (int)((td0 & 0xFFFFu) << 16);               // dim0[15:0] @ bit48
  g1[2] = (int)((td0 >> 16) | ((td1 & 0xFFFFu) << 16));
  g1[3] = (int)((td1 >> 16) | (tile0 << 16));         // tile_dim0 @ bit112
  g1[4] = (int)tile1;                                 // tile_dim1; tile_dim2=0
  g1[5] = (int)(s0 & 0xFFFFFFFFull);                  // dim0_stride[31:0]
  g1[6] = (int)((s0 >> 32) & 0xFFFFull);              // dim0_stride[47:32]
  g1[7] = 0;                                          // dim1_stride hi = 0
  i32x4 z4 = {0, 0, 0, 0};
  i32x8 z8 = {0, 0, 0, 0, 0, 0, 0, 0};
  __builtin_amdgcn_tensor_load_to_lds(g0, g1, z4, z4, z8, 0);
}

// ---------------------------------------------------------------------------
// f32 -> bf16 conversion (weights, once per call)
// ---------------------------------------------------------------------------
__global__ void f2bf_kernel(const float* __restrict__ in, bf16* __restrict__ out,
                            int n) {
  for (int i = blockIdx.x * blockDim.x + threadIdx.x; i < n;
       i += gridDim.x * blockDim.x)
    out[i] = (bf16)in[i];
}

// ---------------------------------------------------------------------------
// Embedding lookup * sqrt(D): one block per token, 256 threads x 2 elems
// ---------------------------------------------------------------------------
__global__ __launch_bounds__(256) void embed_kernel(
    const long long* __restrict__ src, const float* __restrict__ emb,
    float* __restrict__ h, bf16* __restrict__ xb) {
  int t = blockIdx.x;
  long long tok = src[t];
  int i = threadIdx.x * 2;
  size_t base = (size_t)t * DMODEL;
  const float sc = 22.62741699796952f;       // sqrt(512)
  float v0 = emb[(size_t)tok * DMODEL + i] * sc;
  float v1 = emb[(size_t)tok * DMODEL + i + 1] * sc;
  h[base + i]      = v0;  h[base + i + 1] = v1;
  xb[base + i]     = (bf16)v0;
  xb[base + i + 1] = (bf16)v1;
}

// ---------------------------------------------------------------------------
// Fused Q/K/V projection. x viewed as [N*S*H, 64]; W is 64x64 shared by heads.
// Q,K stored [N,H,S,64]; V stored transposed [N,H,64,S] so flash-attention's
// P*V B-fragments are contiguous.
// ---------------------------------------------------------------------------
__device__ __forceinline__ void proj_head(v16bf a0, v16bf a1,
    const bf16* __restrict__ W, const float* __restrict__ bias,
    bf16* __restrict__ out, int transposed, int m0, int lane) {
  v8f acc[4] = {};
#pragma unroll
  for (int ct = 0; ct < 4; ++ct) {
    acc[ct] = wmma_bf16(a0, load_b_frag(W + ct * 16 * HDIM,      HDIM, lane), acc[ct]);
    acc[ct] = wmma_bf16(a1, load_b_frag(W + ct * 16 * HDIM + 32, HDIM, lane), acc[ct]);
  }
  int rowoff = (lane >> 4) << 3;
  int cl = lane & 15;
#pragma unroll
  for (int ct = 0; ct < 4; ++ct) {
    int col = ct * 16 + cl;
    float bv = bias[col];
#pragma unroll
    for (int r = 0; r < 8; ++r) {
      int g = m0 + rowoff + r;               // row = (n*S + s)*H + h
      int n = g >> 14;                       // S*H = 16384
      int rem = g & 16383;
      int s = rem >> 3, hh = rem & 7;        // H = 8
      float v = acc[ct][r] + bv;
      size_t idx = transposed
          ? ((size_t)(n * NHEAD + hh) * HDIM + col) * SEQ + s
          : ((size_t)(n * NHEAD + hh) * SEQ + s) * HDIM + col;
      out[idx] = (bf16)v;
    }
  }
}

__global__ __launch_bounds__(256) void qkv_kernel(
    const bf16* __restrict__ xb,
    const bf16* __restrict__ wq, const bf16* __restrict__ wk,
    const bf16* __restrict__ wv,
    const float* __restrict__ bq, const float* __restrict__ bk,
    const float* __restrict__ bv,
    bf16* __restrict__ Qb, bf16* __restrict__ Kb, bf16* __restrict__ Vt) {
  int wave = threadIdx.x >> 5, lane = threadIdx.x & 31;
  int m0 = blockIdx.x * 128 + wave * 16;
  const bf16* arow = xb + (size_t)m0 * HDIM;
  v16bf a0 = load_a_frag(arow,      HDIM, lane);
  v16bf a1 = load_a_frag(arow + 32, HDIM, lane);
  proj_head(a0, a1, wq, bq, Qb, 0, m0, lane);
  proj_head(a0, a1, wk, bk, Kb, 0, m0, lane);
  proj_head(a0, a1, wv, bv, Vt, 1, m0, lane);
}

// ---------------------------------------------------------------------------
// Flash attention: block = (n, h, 128-query tile); each of 8 waves owns 16
// query rows x all 64 head dims. Online softmax over 64-key tiles; score
// tiles never leave registers/LDS.  P (C-layout) -> LDS -> A-fragments.
// ---------------------------------------------------------------------------
__global__ __launch_bounds__(256) void flash_attn_kernel(
    const bf16* __restrict__ Qb, const bf16* __restrict__ Kb,
    const bf16* __restrict__ Vt, const int* __restrict__ mask,
    bf16* __restrict__ ctx) {
  __shared__ bf16 plds[8 * 16 * 64];         // 16 KB, per-wave private slabs
  const int wave = threadIdx.x >> 5, lane = threadIdx.x & 31;
  int bid = blockIdx.x;
  int qb = bid & 15;                         // S/128 = 16 query tiles
  int h  = (bid >> 4) & 7;
  int n  = bid >> 7;
  const int q0 = qb * 128 + wave * 16;
  const size_t head  = (size_t)(n * NHEAD + h) * SEQ * HDIM;  // Q/K base
  const size_t headV = (size_t)(n * NHEAD + h) * HDIM * SEQ;  // Vt base

  v16bf qf0 = load_a_frag(Qb + head + (size_t)q0 * HDIM,      HDIM, lane);
  v16bf qf1 = load_a_frag(Qb + head + (size_t)q0 * HDIM + 32, HDIM, lane);

  v8f O[4] = {};
  float mrow[8], lrow[8];
#pragma unroll
  for (int r = 0; r < 8; ++r) { mrow[r] = -1e30f; lrow[r] = 0.f; }

  const float scale = 0.04419417382415922f;  // 1/sqrt(512) (ref scales by D!)
  bf16* pw = plds + wave * (16 * 64);
  const int rowoff = (lane >> 4) << 3;
  const int cl = lane & 15;

  for (int kb = 0; kb < SEQ; kb += 64) {
    // ---- S = Q * K^T for a 16q x 64k tile -------------------------------
    v8f Sv[4];
#pragma unroll
    for (int ct = 0; ct < 4; ++ct) {
      v8f z = {0.f, 0.f, 0.f, 0.f, 0.f, 0.f, 0.f, 0.f};
      const bf16* pk = Kb + head + (size_t)(kb + ct * 16) * HDIM;
      z = wmma_bf16(qf0, load_b_frag(pk,      HDIM, lane), z);
      z = wmma_bf16(qf1, load_b_frag(pk + 32, HDIM, lane), z);
      Sv[ct] = z;
    }
    // ---- mask (before scale, as in reference) + scale -------------------
#pragma unroll
    for (int ct = 0; ct < 4; ++ct) {
      int mv = mask[n * SEQ + kb + ct * 16 + cl];
#pragma unroll
      for (int r = 0; r < 8; ++r) {
        float e = Sv[ct][r];
        if (mv == 0) e = -1e10f;
        Sv[ct][r] = e * scale;
      }
    }
    // ---- online softmax: row max over 64 keys (16-lane shuffle groups) --
    float mn[8];
#pragma unroll
    for (int r = 0; r < 8; ++r) {
      float lm = fmaxf(fmaxf(Sv[0][r], Sv[1][r]), fmaxf(Sv[2][r], Sv[3][r]));
#pragma unroll
      for (int off = 1; off < 16; off <<= 1)
        lm = fmaxf(lm, __shfl_xor(lm, off, 16));
      mn[r] = fmaxf(mrow[r], lm);
    }
    float rs[8];
#pragma unroll
    for (int r = 0; r < 8; ++r) {
      float acc = 0.f;
#pragma unroll
      for (int ct = 0; ct < 4; ++ct) {
        float p = __expf(Sv[ct][r] - mn[r]);
        Sv[ct][r] = p;
        acc += p;
      }
#pragma unroll
      for (int off = 1; off < 16; off <<= 1) acc += __shfl_xor(acc, off, 16);
      rs[r] = acc;
    }
#pragma unroll
    for (int r = 0; r < 8; ++r) {
      float al = __expf(mrow[r] - mn[r]);
      lrow[r] = lrow[r] * al + rs[r];
      mrow[r] = mn[r];
#pragma unroll
      for (int ct = 0; ct < 4; ++ct) O[ct][r] *= al;
    }
    // ---- P (C layout) -> LDS -> A-fragment layout -----------------------
#pragma unroll
    for (int ct = 0; ct < 4; ++ct)
#pragma unroll
      for (int r = 0; r < 8; ++r)
        pw[(r + rowoff) * 64 + ct * 16 + cl] = (bf16)Sv[ct][r];
    asm volatile("s_wait_dscnt 0" ::: "memory");  // cross-lane LDS RAW
    v16bf p0 = load_a_frag(pw,      64, lane);
    v16bf p1 = load_a_frag(pw + 32, 64, lane);
    // ---- O += P * V  (Vt is [64, S]: contiguous B-fragments) ------------
#pragma unroll
    for (int ct = 0; ct < 4; ++ct) {
      const bf16* pv = Vt + headV + (size_t)(ct * 16) * SEQ + kb;
      O[ct] = wmma_bf16(p0, load_b_frag(pv,      SEQ, lane), O[ct]);
      O[ct] = wmma_bf16(p1, load_b_frag(pv + 32, SEQ, lane), O[ct]);
    }
  }
  // ---- normalize and write context in [N,S,D] bf16 ----------------------
#pragma unroll
  for (int r = 0; r < 8; ++r) {
    float inv = 1.f / fmaxf(lrow[r], 1e-20f);
    int grow = q0 + rowoff + r;
#pragma unroll
    for (int ct = 0; ct < 4; ++ct) {
      float v = O[ct][r] * inv;
      ctx[((size_t)(n * SEQ + grow)) * DMODEL + h * HDIM + ct * 16 + cl] = (bf16)v;
    }
  }
}

// ---------------------------------------------------------------------------
// Generic GEMM: C[M,N] = act(A[M,K] * W[N,K]^T + bias). 128x64 tile/block,
// 8 waves, each wave = 16 rows x 64 cols (4 WMMA col tiles), K-step 32.
// The shared 64x32 weight tile is DMA'd into LDS by the Tensor Data Mover
// (one descriptor per block per k-step, double-buffered): removes the 8x
// redundant per-wave global B loads and overlaps DMA with WMMA.
// ---------------------------------------------------------------------------
template <bool RELU, bool OUT_BF16>
__global__ __launch_bounds__(256) void gemm_kernel(
    const bf16* __restrict__ A, const bf16* __restrict__ W,
    const float* __restrict__ bias, float* __restrict__ Cf,
    bf16* __restrict__ Cb, int M, int N, int K) {
  __shared__ bf16 bsm[2][64 * 32];           // double-buffered B tile (8 KB)
  int wave = threadIdx.x >> 5, lane = threadIdx.x & 31;
  int m0 = blockIdx.x * 128 + wave * 16;
  int n0 = blockIdx.y * 64;
  const bf16* wbase = W + (size_t)n0 * K;
  unsigned lds0 = (unsigned)(size_t)&bsm[0][0];  // generic low 32 = LDS offset
  unsigned lds1 = (unsigned)(size_t)&bsm[1][0];

  if (wave == 0) {                            // one TDM issue per block
    tdm_load_tile64x32(lds0, wbase, K);
    __builtin_amdgcn_s_wait_tensorcnt(0);
  }
  __syncthreads();

  v8f acc[4] = {};
  const bf16* arow = A + (size_t)m0 * K;
  int nk = K >> 5;
  for (int kk = 0; kk < nk; ++kk) {
    if (wave == 0 && kk + 1 < nk)             // prefetch next tile via TDM
      tdm_load_tile64x32((kk & 1) ? lds0 : lds1, wbase + (kk + 1) * 32, K);
    __builtin_prefetch(arow + ((lane & 15) * (size_t)K) + kk * 32 + 128, 0, 1);
    const bf16* bt = &bsm[kk & 1][0];
    v16bf a = load_a_frag(arow + kk * 32, K, lane);
#pragma unroll
    for (int ct = 0; ct < 4; ++ct)
      acc[ct] = wmma_bf16(a, load_b_frag_lds(bt + ct * 16 * 32, lane), acc[ct]);
    if (wave == 0) __builtin_amdgcn_s_wait_tensorcnt(0);
    __syncthreads();                          // publish next buffer / WAR fence
  }

  int rowoff = (lane >> 4) << 3;
  int cl = lane & 15;
#pragma unroll
  for (int ct = 0; ct < 4; ++ct) {
    int col = n0 + ct * 16 + cl;
    float bv = bias[col];
#pragma unroll
    for (int r = 0; r < 8; ++r) {
      size_t idx = (size_t)(m0 + rowoff + r) * N + col;
      float v = acc[ct][r] + bv;
      if (RELU) v = fmaxf(v, 0.f);
      if (OUT_BF16) Cb[idx] = (bf16)v;
      else          Cf[idx] = v;
    }
  }
}

// ---------------------------------------------------------------------------
// Residual add + LayerNorm: one block per token row (D=512, 2 elems/thread).
// Writes f32 hidden state (in place into h) and a bf16 mirror for GEMMs.
// ---------------------------------------------------------------------------
__global__ __launch_bounds__(256) void add_ln_kernel(
    const float* __restrict__ hin, const float* __restrict__ y,
    const float* __restrict__ g, const float* __restrict__ b,
    float* __restrict__ hout, bf16* __restrict__ xb) {
  __shared__ float red[16];
  int t = blockIdx.x;
  int i = threadIdx.x * 2;
  size_t base = (size_t)t * DMODEL;
  float x0 = hin[base + i]     + y[base + i];
  float x1 = hin[base + i + 1] + y[base + i + 1];
  float s  = x0 + x1;
  float ss = x0 * x0 + x1 * x1;
#pragma unroll
  for (int off = 1; off < 32; off <<= 1) {
    s  += __shfl_xor(s,  off, 32);
    ss += __shfl_xor(ss, off, 32);
  }
  int wave = threadIdx.x >> 5, lane = threadIdx.x & 31;
  if (lane == 0) { red[wave] = s; red[8 + wave] = ss; }
  __syncthreads();
  float tot = 0.f, tot2 = 0.f;
#pragma unroll
  for (int w = 0; w < 8; ++w) { tot += red[w]; tot2 += red[8 + w]; }
  float mean = tot * (1.f / DMODEL);
  float var  = tot2 * (1.f / DMODEL) - mean * mean;
  float rstd = rsqrtf(var + 1e-5f);
  float o0 = (x0 - mean) * rstd * g[i]     + b[i];
  float o1 = (x1 - mean) * rstd * g[i + 1] + b[i + 1];
  hout[base + i]     = o0;  hout[base + i + 1] = o1;
  xb[base + i]       = (bf16)o0;
  xb[base + i + 1]   = (bf16)o1;
}

// ---------------------------------------------------------------------------
// Host-side orchestration
// ---------------------------------------------------------------------------
extern "C" void kernel_launch(void* const* d_in, const int* in_sizes, int n_in,
                              void* d_out, int out_size, void* d_ws,
                              size_t ws_size, hipStream_t stream) {
  const long long* src  = (const long long*)d_in[0];
  const int*   mask     = (const int*)d_in[1];
  const float* emb      = (const float*)d_in[2];
  const float* Wq = (const float*)d_in[3];  const float* bq = (const float*)d_in[4];
  const float* Wk = (const float*)d_in[5];  const float* bk = (const float*)d_in[6];
  const float* Wv = (const float*)d_in[7];  const float* bv = (const float*)d_in[8];
  const float* Wo = (const float*)d_in[9];  const float* bo = (const float*)d_in[10];
  const float* W1 = (const float*)d_in[11]; const float* b1 = (const float*)d_in[12];
  const float* W2 = (const float*)d_in[13]; const float* b2 = (const float*)d_in[14];
  const float* ln1g = (const float*)d_in[15]; const float* ln1b = (const float*)d_in[16];
  const float* ln2g = (const float*)d_in[17]; const float* ln2b = (const float*)d_in[18];

  float* h = (float*)d_out;                       // f32 hidden state [N,S,D]

  // Workspace bump allocator (256B aligned); ~75 MB total.
  char* ws = (char*)d_ws;
  auto alloc = [&](size_t bytes) -> char* {
    char* p = ws;
    ws += (bytes + 255) & ~(size_t)255;
    return p;
  };
  const size_t NSD = (size_t)NTOK * DMODEL;       // 2,097,152
  bf16* xb   = (bf16*)alloc(NSD * 2);             // bf16 hidden mirror
  bf16* Qb   = (bf16*)alloc(NSD * 2);
  bf16* Kb   = (bf16*)alloc(NSD * 2);
  bf16* Vt   = (bf16*)alloc(NSD * 2);             // V transposed [N,H,64,S]
  bf16* ctx  = (bf16*)alloc(NSD * 2);
  float* y   = (float*)alloc(NSD * 4);            // GEMM f32 output
  bf16* ff1  = (bf16*)alloc((size_t)NTOK * FDIM * 2);
  bf16* wq_b = (bf16*)alloc((size_t)LNUM * HDIM * HDIM * 2);
  bf16* wk_b = (bf16*)alloc((size_t)LNUM * HDIM * HDIM * 2);
  bf16* wv_b = (bf16*)alloc((size_t)LNUM * HDIM * HDIM * 2);
  bf16* wo_b = (bf16*)alloc((size_t)LNUM * DMODEL * DMODEL * 2);
  bf16* w1_b = (bf16*)alloc((size_t)LNUM * FDIM * DMODEL * 2);
  bf16* w2_b = (bf16*)alloc((size_t)LNUM * DMODEL * FDIM * 2);

  // Weight conversion (cheap; repeated every call for determinism).
  f2bf_kernel<<<256,  256, 0, stream>>>(Wq, wq_b, LNUM * HDIM * HDIM);
  f2bf_kernel<<<256,  256, 0, stream>>>(Wk, wk_b, LNUM * HDIM * HDIM);
  f2bf_kernel<<<256,  256, 0, stream>>>(Wv, wv_b, LNUM * HDIM * HDIM);
  f2bf_kernel<<<2048, 256, 0, stream>>>(Wo, wo_b, LNUM * DMODEL * DMODEL);
  f2bf_kernel<<<4096, 256, 0, stream>>>(W1, w1_b, LNUM * FDIM * DMODEL);
  f2bf_kernel<<<4096, 256, 0, stream>>>(W2, w2_b, LNUM * DMODEL * FDIM);

  // Embedding * sqrt(D)
  embed_kernel<<<NTOK, 256, 0, stream>>>(src, emb, h, xb);

  for (int l = 0; l < LNUM; ++l) {
    // QKV projections (shared 64x64 per-head weights)
    qkv_kernel<<<NROWQ / 128, 256, 0, stream>>>(
        xb, wq_b + (size_t)l * HDIM * HDIM, wk_b + (size_t)l * HDIM * HDIM,
        wv_b + (size_t)l * HDIM * HDIM, bq + l * HDIM, bk + l * HDIM,
        bv + l * HDIM, Qb, Kb, Vt);
    // Flash attention: N*H*(S/128) = 256 blocks
    flash_attn_kernel<<<NB * NHEAD * (SEQ / 128), 256, 0, stream>>>(
        Qb, Kb, Vt, mask, ctx);
    // Output projection Wo -> y (f32)
    gemm_kernel<false, false><<<dim3(NTOK / 128, DMODEL / 64), 256, 0, stream>>>(
        ctx, wo_b + (size_t)l * DMODEL * DMODEL, bo + (size_t)l * DMODEL,
        y, nullptr, NTOK, DMODEL, DMODEL);
    // h = LN1(h + y)
    add_ln_kernel<<<NTOK, 256, 0, stream>>>(h, y, ln1g + (size_t)l * DMODEL,
                                            ln1b + (size_t)l * DMODEL, h, xb);
    // FFN1 with ReLU -> bf16
    gemm_kernel<true, true><<<dim3(NTOK / 128, FDIM / 64), 256, 0, stream>>>(
        xb, w1_b + (size_t)l * FDIM * DMODEL, b1 + (size_t)l * FDIM,
        nullptr, ff1, NTOK, FDIM, DMODEL);
    // FFN2 -> y (f32)
    gemm_kernel<false, false><<<dim3(NTOK / 128, DMODEL / 64), 256, 0, stream>>>(
        ff1, w2_b + (size_t)l * DMODEL * FDIM, b2 + (size_t)l * DMODEL,
        y, nullptr, NTOK, DMODEL, FDIM);
    // h = LN2(h + y)
    add_ln_kernel<<<NTOK, 256, 0, stream>>>(h, y, ln2g + (size_t)l * DMODEL,
                                            ln2b + (size_t)l * DMODEL, h, xb);
  }
  (void)in_sizes; (void)n_in; (void)out_size; (void)ws_size;
}
